// QuantizedLinear_16595753632078
// MI455X (gfx1250) — 8-bit matrix operand order repaired and checked by reading, compiled, not run on hardware
//
#include <hip/hip_runtime.h>
#include <hip/hip_bf16.h>

typedef __attribute__((ext_vector_type(16))) int   v16i;
typedef __attribute__((ext_vector_type(8)))  float v8f;
typedef int v4i_b __attribute__((vector_size(16)));

#define K_DIM   4096
#define N_DIM   4096
#define M_DIM   2048
#define NGROUPS 32
#define LDSS    144

#if __has_builtin(__builtin_amdgcn_global_load_async_to_lds_b128)
#define QL_ASYNC 1
#define CP16(ldsp, gp)                                                        \
    __builtin_amdgcn_global_load_async_to_lds_b128(                           \
        (__attribute__((address_space(1))) v4i_b*)(v4i_b*)(void*)(gp),        \
        (__attribute__((address_space(3))) v4i_b*)(v4i_b*)(void*)(ldsp), 0, 0)
#else
#define QL_ASYNC 0
#define CP16(ldsp, gp) (*(uint4*)(ldsp) = *(const uint4*)(gp))
#endif

#if QL_ASYNC && __has_builtin(__builtin_amdgcn_s_wait_asynccnt)
#define WAIT_ASYNC(n) __builtin_amdgcn_s_wait_asynccnt(n)
#elif QL_ASYNC
#define WAIT_ASYNC(n) asm volatile("s_wait_asynccnt %0" ::"i"(n))
#else
#define WAIT_ASYNC(n) ((void)0)
#endif

__device__ __forceinline__ unsigned int fp4_grid_as_fp8(float r) {
    float a = fabsf(r);
    unsigned int c =
        a < 0.25f ? 0x00u :
        a < 0.75f ? 0x30u :
        a < 1.25f ? 0x38u :
        a < 1.75f ? 0x3Cu :
        a < 2.50f ? 0x40u :
        a < 3.50f ? 0x44u :
        a < 5.00f ? 0x48u : 0x4Cu;
    return c | (r < 0.f ? 0x80u : 0x00u);
}

__global__ __launch_bounds__(256)
void ql_quant_x(const float* __restrict__ x, unsigned char* __restrict__ xq,
                float* __restrict__ xs) {
    __shared__ float red[8];
    const int row = blockIdx.x;
    const int t   = threadIdx.x;
    const float* xr = x + (size_t)row * K_DIM + t * 16;

    float v[16];
    float m = 0.f;
#pragma unroll
    for (int i = 0; i < 4; ++i) {
        float4 d = ((const float4*)xr)[i];
        v[4*i+0] = fminf(fmaxf(d.x, -3.f), 3.f);
        v[4*i+1] = fminf(fmaxf(d.y, -3.f), 3.f);
        v[4*i+2] = fminf(fmaxf(d.z, -3.f), 3.f);
        v[4*i+3] = fminf(fmaxf(d.w, -3.f), 3.f);
        m = fmaxf(m, fmaxf(fmaxf(fabsf(v[4*i+0]), fabsf(v[4*i+1])),
                           fmaxf(fabsf(v[4*i+2]), fabsf(v[4*i+3]))));
    }
#pragma unroll
    for (int off = 16; off; off >>= 1) m = fmaxf(m, __shfl_xor(m, off, 32));
    if ((t & 31) == 0) red[t >> 5] = m;
    __syncthreads();
    float rm = red[0];
#pragma unroll
    for (int i = 1; i < 8; ++i) rm = fmaxf(rm, red[i]);

    const float scale = fmaxf(rm, 1e-30f) * (1.f / 6.f);
    const float inv   = 1.f / scale;
    if (t == 0) xs[row] = scale;

    unsigned int packed[4];
#pragma unroll
    for (int i = 0; i < 4; ++i) {
        unsigned int b0 = fp4_grid_as_fp8(v[4*i+0] * inv);
        unsigned int b1 = fp4_grid_as_fp8(v[4*i+1] * inv);
        unsigned int b2 = fp4_grid_as_fp8(v[4*i+2] * inv);
        unsigned int b3 = fp4_grid_as_fp8(v[4*i+3] * inv);
        packed[i] = b0 | (b1 << 8) | (b2 << 16) | (b3 << 24);
    }
    uint4* dst = (uint4*)(xq + (size_t)row * K_DIM + t * 16);
    *dst = make_uint4(packed[0], packed[1], packed[2], packed[3]);
}

__global__ __launch_bounds__(256)
void ql_quant_w(const float* __restrict__ w, unsigned char* __restrict__ wq,
                float* __restrict__ ws) {
    const int row = blockIdx.x;
    const int t   = threadIdx.x;
    const float* wr = w + (size_t)row * K_DIM + t * 16;

    float v[16];
    float m = 0.f;
#pragma unroll
    for (int i = 0; i < 4; ++i) {
        float4 d = ((const float4*)wr)[i];
        v[4*i+0] = d.x; v[4*i+1] = d.y; v[4*i+2] = d.z; v[4*i+3] = d.w;
        m = fmaxf(m, fmaxf(fmaxf(fabsf(d.x), fabsf(d.y)),
                           fmaxf(fabsf(d.z), fabsf(d.w))));
    }
#pragma unroll
    for (int off = 4; off; off >>= 1) m = fmaxf(m, __shfl_xor(m, off, 32));

    const float scale = fmaxf(m, 1e-30f) * (1.f / 6.f);
    const float inv   = 1.f / scale;
    if ((t & 7) == 0) ws[(size_t)row * NGROUPS + (t >> 3)] = scale;

    unsigned int packed[4];
#pragma unroll
    for (int i = 0; i < 4; ++i) {
        unsigned int b0 = fp4_grid_as_fp8(v[4*i+0] * inv);
        unsigned int b1 = fp4_grid_as_fp8(v[4*i+1] * inv);
        unsigned int b2 = fp4_grid_as_fp8(v[4*i+2] * inv);
        unsigned int b3 = fp4_grid_as_fp8(v[4*i+3] * inv);
        packed[i] = b0 | (b1 << 8) | (b2 << 16) | (b3 << 24);
    }
    uint4* dst = (uint4*)(wq + (size_t)row * K_DIM + t * 16);
    *dst = make_uint4(packed[0], packed[1], packed[2], packed[3]);
}

__global__ __launch_bounds__(256)
void ql_gemm_fp8(const unsigned char* __restrict__ xq,
                 const unsigned char* __restrict__ wq,
                 const float* __restrict__ xs,
                 const float* __restrict__ ws,
                 const float* __restrict__ bias,
                 float* __restrict__ out) {
    __shared__ __align__(16) unsigned char smA[2][64  * LDSS];
    __shared__ __align__(16) unsigned char smB[2][128 * LDSS];

    const int tid   = threadIdx.x;
    const int lane  = tid & 31;
    const int wave  = tid >> 5;
    const int half  = lane >> 4;
    const int l15   = lane & 15;
    const int waveM = wave >> 1;
    const int waveN = wave & 1;
    const int mBlock = blockIdx.y * 64;
    const int nBlock = blockIdx.x * 128;

    auto stage = [&](int g, int buf) {
        const int kOff = g * 128;
#pragma unroll
        for (int c = 0; c < 2; ++c) {
            const int i   = tid + c * 256;
            const int row = i >> 3;
            const int kb  = (i & 7) * 16;
            CP16(&smA[buf][row * LDSS + kb],
                 xq + (size_t)(mBlock + row) * K_DIM + kOff + kb);
        }
#pragma unroll
        for (int c = 0; c < 4; ++c) {
            const int i   = tid + c * 256;
            const int col = i >> 3;
            const int kb  = (i & 7) * 16;
            CP16(&smB[buf][col * LDSS + kb],
                 wq + (size_t)(nBlock + col) * K_DIM + kOff + kb);
        }
    };

    float acc[4][8];
#pragma unroll
    for (int nb = 0; nb < 4; ++nb)
#pragma unroll
        for (int r = 0; r < 8; ++r) acc[nb][r] = 0.f;

    stage(0, 0);

    for (int g = 0; g < NGROUPS; ++g) {
        const int buf = g & 1;
        if (g + 1 < NGROUPS) {
            stage(g + 1, buf ^ 1);
            WAIT_ASYNC(6);
        } else {
            WAIT_ASYNC(0);
        }
        __syncthreads();

        const unsigned char* aP = &smA[buf][(waveM * 16 + l15) * LDSS + half * 16];
        v16i A;
#pragma unroll
        for (int p = 0; p < 8; ++p) {
            uint2 d = *(const uint2*)(aP + (p >> 1) * 32 + (p & 1) * 8);
            A[2*p]   = (int)d.x;
            A[2*p+1] = (int)d.y;
        }

#pragma unroll
        for (int nb = 0; nb < 4; ++nb) {
            const int cLoc = waveN * 64 + nb * 16 + l15;
            const unsigned char* bP = &smB[buf][cLoc * LDSS + half * 16];
            v16i B;
#pragma unroll
            for (int q = 0; q < 4; ++q) {
                uint4 d = *(const uint4*)(bP + q * 32);
                B[4*q]   = (int)d.x;
                B[4*q+1] = (int)d.y;
                B[4*q+2] = (int)d.z;
                B[4*q+3] = (int)d.w;
            }
            v8f c = {};
            v8f dv = __builtin_amdgcn_wmma_f32_16x16x128_fp8_fp8(
                A, B, (short)0, c, false, false);
            const float s = ws[(size_t)(nBlock + cLoc) * NGROUPS + g];
#pragma unroll
            for (int r = 0; r < 8; ++r)
                acc[nb][r] = fmaf(s, dv[r], acc[nb][r]);
        }
        __syncthreads();
    }

    const int mBase = mBlock + waveM * 16;
    float sxr[8];
#pragma unroll
    for (int r = 0; r < 8; ++r) sxr[r] = xs[mBase + half * 8 + r];
#pragma unroll
    for (int nb = 0; nb < 4; ++nb) {
        const int n  = nBlock + waveN * 64 + nb * 16 + l15;
        const float b = bias[n];
#pragma unroll
        for (int r = 0; r < 8; ++r) {
            const int m = mBase + half * 8 + r;
            out[(size_t)m * N_DIM + n] = fmaf(sxr[r], acc[nb][r], b);
        }
    }
}

extern "C" void kernel_launch(void* const* d_in, const int* in_sizes, int n_in,
                              void* d_out, int out_size, void* d_ws, size_t ws_size,
                              hipStream_t stream) {
    const float* x    = (const float*)d_in[0];
    const float* w    = (const float*)d_in[1];
    const float* bias = (const float*)d_in[2];
    float* out = (float*)d_out;

    unsigned char* xq = (unsigned char*)d_ws;
    unsigned char* wq = xq + (size_t)M_DIM * K_DIM;
    float* xs = (float*)(wq + (size_t)N_DIM * K_DIM);
    float* ws = xs + M_DIM;

    ql_quant_x<<<M_DIM, 256, 0, stream>>>(x, xq, xs);
    ql_quant_w<<<N_DIM, 256, 0, stream>>>(w, wq, ws);

    dim3 grid(N_DIM / 128, M_DIM / 64);
    ql_gemm_fp8<<<grid, 256, 0, stream>>>(xq, wq, xs, ws, bias, out);
}
